// CustomLongcatMoe_68917045231896
// MI455X (gfx1250) — compile-verified, matrix operand order checked by reading, not run on hardware
//
#include <hip/hip_runtime.h>
#include <hip/hip_bf16.h>
#include <math.h>

// Problem dims (fixed by reference)
#define TOKENS 4096
#define HID    1024
#define INTER  512
#define NEXP   16
#define NSLOT  32
#define TOPK   4

// Tiling
#define BM 64        // tokens per block tile
#define BNT 64       // output cols per block tile (per matrix)
#define BK 64        // K per pipeline stage (2 WMMA K-steps)
#define LPAD 8
#define LW (BK + LPAD)   // 72 bf16 = 144B row stride: 16B aligned, conflict-free

typedef __attribute__((ext_vector_type(16))) __bf16 v16bf;
typedef __attribute__((ext_vector_type(8)))  __bf16 v8bf;
typedef __attribute__((ext_vector_type(8)))  float  v8f;

union ABFrag { v16bf v; v8bf h[2]; };

// ---------------------------------------------------------------------------
// CDNA5 async global->LDS copy (ASYNCcnt-tracked, bypasses VGPRs)
// ---------------------------------------------------------------------------
__device__ __forceinline__ unsigned lds_lo(const void* p) {
  // flat shared pointer: addr[31:0] is the LDS address (aperture in high bits)
  return (unsigned)(size_t)p;
}
__device__ __forceinline__ void async_copy16(unsigned lds, const __bf16* g) {
  asm volatile("global_load_async_to_lds_b128 %0, %1, off"
               :: "v"(lds), "v"(g) : "memory");
}
__device__ __forceinline__ void async_wait0() {
  asm volatile("s_wait_asynccnt 0" ::: "memory");
}

// ---------------------------------------------------------------------------
// fp32 -> bf16 linear conversion (x is consumed row-major as the A operand)
// ---------------------------------------------------------------------------
__global__ __launch_bounds__(256)
void f32_to_bf16_kernel(const float* __restrict__ in, __bf16* __restrict__ out, int n) {
  int i = (blockIdx.x * blockDim.x + threadIdx.x) * 4;
  if (i + 3 < n) {
    float4 v = *(const float4*)(in + i);
    out[i + 0] = (__bf16)v.x;
    out[i + 1] = (__bf16)v.y;
    out[i + 2] = (__bf16)v.z;
    out[i + 3] = (__bf16)v.w;
  } else {
    for (; i < n; ++i) out[i] = (__bf16)in[i];
  }
}

// ---------------------------------------------------------------------------
// fp32 [e][K][N] -> bf16 [e][N][K] (K-contiguous B-operand layout), LDS-tiled
// so both global sides stay coalesced. One-time pass; afterwards every GEMM
// tile load (A and B) is a contiguous 16B async copy.
// ---------------------------------------------------------------------------
__global__ __launch_bounds__(256)
void transpose_f32_to_bf16(const float* __restrict__ in, __bf16* __restrict__ out,
                           int K, int N) {
  __shared__ float t[32][33];
  const int e = blockIdx.z;
  in  += (size_t)e * K * N;
  out += (size_t)e * K * N;
  const int k0 = blockIdx.y * 32;
  const int n0 = blockIdx.x * 32;
  const int tx = threadIdx.x & 31;
  const int ty = threadIdx.x >> 5;          // 32 x 8
#pragma unroll
  for (int j = 0; j < 32; j += 8)
    t[ty + j][tx] = in[(size_t)(k0 + ty + j) * N + n0 + tx];
  __syncthreads();
#pragma unroll
  for (int j = 0; j < 32; j += 8)
    out[(size_t)(n0 + ty + j) * K + k0 + tx] = (__bf16)t[tx][ty + j];
}

// ---------------------------------------------------------------------------
// Router: one wave32 per token, lane == router slot (32 slots).
// ---------------------------------------------------------------------------
__global__ __launch_bounds__(256)
void router_kernel(const float* __restrict__ x,
                   const float* __restrict__ rw,
                   const float* __restrict__ bias,
                   float* __restrict__ cw_real,   // [T, NEXP]
                   float* __restrict__ zero_w) {  // [T]
  const int token = blockIdx.x * 8 + (threadIdx.x >> 5);
  const int lane  = threadIdx.x & 31;
  if (token >= TOKENS) return;

  const float* xr = x + (size_t)token * HID;
  const float* wr = rw + (size_t)lane * HID;
  float acc = 0.f;
  for (int h = 0; h < HID; h += 4) {
    float4 xv = *(const float4*)(xr + h);
    float4 wv = *(const float4*)(wr + h);
    acc += xv.x * wv.x + xv.y * wv.y + xv.z * wv.z + xv.w * wv.w;
  }

  float m = acc;
  for (int off = 16; off; off >>= 1) m = fmaxf(m, __shfl_xor(m, off, 32));
  float ex = __expf(acc - m);
  float s = ex;
  for (int off = 16; off; off >>= 1) s += __shfl_xor(s, off, 32);
  float score = ex / s;

  float key = score + bias[lane];
  int selected = 0;
  for (int k = 0; k < TOPK; ++k) {
    float mv = key; int mi = lane;
    for (int off = 16; off; off >>= 1) {
      float ov = __shfl_xor(mv, off, 32);
      int   oi = __shfl_xor(mi, off, 32);
      if (ov > mv || (ov == mv && oi < mi)) { mv = ov; mi = oi; }
    }
    if (lane == mi) { selected = 1; key = -INFINITY; }
  }

  float w = selected ? score : 0.f;
  if (lane < NEXP) cw_real[(size_t)token * NEXP + lane] = w;

  float zw = (lane >= NEXP) ? w : 0.f;
  for (int off = 16; off; off >>= 1) zw += __shfl_xor(zw, off, 32);
  if (lane == 0) zero_w[token] = zw;
}

// ---------------------------------------------------------------------------
// Gate/Up GEMMs + fused SwiGLU. Double-buffered async-LDS pipeline.
// Block tile 64x64 (per matrix); 8 waves in 4x2, each wave 16x32 per matrix:
// 8 WMMAs / wave / stage, each A fragment feeds 4 WMMAs.
// ---------------------------------------------------------------------------
__global__ __launch_bounds__(256)
void moe_gateup_kernel(const __bf16* __restrict__ xb,    // [T][H]
                       const __bf16* __restrict__ wgt,   // [E][I][H] (transposed)
                       const __bf16* __restrict__ wut,   // [E][I][H]
                       __bf16* __restrict__ act) {       // [E][T][I]
  __shared__ __bf16 lx[2][BM][LW];
  __shared__ __bf16 lg[2][BNT][LW];
  __shared__ __bf16 lu[2][BNT][LW];

  const int e    = blockIdx.z;
  const int m0   = blockIdx.y * BM;
  const int n0   = blockIdx.x * BNT;
  const int tid  = threadIdx.x;
  const int wid  = tid >> 5;
  const int lane = tid & 31;
  const int half = lane >> 4;
  const int ml   = lane & 15;
  const int wm   = (wid & 3) * 16;     // wave M origin
  const int wn   = (wid >> 2) * 32;    // wave N origin (2 x 16)

  const __bf16* xe  = xb + (size_t)m0 * HID;
  const __bf16* wge = wgt + (size_t)e * INTER * HID + (size_t)n0 * HID;
  const __bf16* wue = wut + (size_t)e * INTER * HID + (size_t)n0 * HID;

  const int crow = tid >> 3;           // copy coords: 8 chunks per 64-wide row
  const int ccg  = (tid & 7) * 8;

  auto issue = [&](int s, int b) {
    const int k0 = s * BK;
#pragma unroll
    for (int j = 0; j < 2; ++j) {      // each tile 64x64 = 512 chunks, 2/thread
      const int row = crow + j * 32;
      async_copy16(lds_lo(&lx[b][row][ccg]), xe  + (size_t)row * HID + k0 + ccg);
      async_copy16(lds_lo(&lg[b][row][ccg]), wge + (size_t)row * HID + k0 + ccg);
      async_copy16(lds_lo(&lu[b][row][ccg]), wue + (size_t)row * HID + k0 + ccg);
    }
  };

  constexpr int NSTAGE = HID / BK;     // 16
  v8f accg[2] = {};
  v8f accu[2] = {};

  issue(0, 0);
  for (int s = 0; s < NSTAGE; ++s) {
    async_wait0();
    __syncthreads();
    if (s + 1 < NSTAGE) issue(s + 1, (s + 1) & 1);
    if (s + 2 < NSTAGE)
      __builtin_prefetch(xe + (size_t)crow * HID + (s + 2) * BK + ccg, 0, 3);

    const int b = s & 1;
#pragma unroll
    for (int kk = 0; kk < 2; ++kk) {
      const int ko = 32 * kk;
      ABFrag a;
      a.h[0] = *(const v8bf*)&lx[b][wm + ml][ko + 8 * half];
      a.h[1] = *(const v8bf*)&lx[b][wm + ml][ko + 16 + 8 * half];
#pragma unroll
      for (int nf = 0; nf < 2; ++nf) {
        const int nr = wn + nf * 16 + ml;
        ABFrag bg, bu;
        bg.h[0] = *(const v8bf*)&lg[b][nr][ko + 8 * half];
        bg.h[1] = *(const v8bf*)&lg[b][nr][ko + 16 + 8 * half];
        bu.h[0] = *(const v8bf*)&lu[b][nr][ko + 8 * half];
        bu.h[1] = *(const v8bf*)&lu[b][nr][ko + 16 + 8 * half];
        accg[nf] = __builtin_amdgcn_wmma_f32_16x16x32_bf16(
            false, a.v, false, bg.v, (short)0, accg[nf], false, false);
        accu[nf] = __builtin_amdgcn_wmma_f32_16x16x32_bf16(
            false, a.v, false, bu.v, (short)0, accu[nf], false, false);
      }
    }
  }

  // Fused SwiGLU; C layout: VGPR r -> M = r + 8*half, N = ml
  __bf16* arow = act + (size_t)e * TOKENS * INTER;
#pragma unroll
  for (int nf = 0; nf < 2; ++nf) {
#pragma unroll
    for (int r = 0; r < 8; ++r) {
      float g = accg[nf][r];
      float u = accu[nf][r];
      float sv = g / (1.0f + __expf(-g));
      int m = m0 + wm + r + 8 * half;
      int n = n0 + wn + nf * 16 + ml;
      arow[(size_t)m * INTER + n] = (__bf16)(sv * u);
    }
  }
}

// ---------------------------------------------------------------------------
// Down-proj + weighted combine. Single pipeline over (expert, k-stage) so the
// async double-buffer never drains across expert boundaries. No atomics ->
// bitwise-deterministic register accumulation across all 16 experts.
// ---------------------------------------------------------------------------
__global__ __launch_bounds__(256)
void moe_down_kernel(const __bf16* __restrict__ act,   // [E][T][I]
                     const __bf16* __restrict__ wdt,   // [E][H][I] (transposed)
                     const float* __restrict__ cw,     // [T][NEXP]
                     const float* __restrict__ zw,     // [T]
                     const float* __restrict__ x,      // [T][H] fp32
                     float* __restrict__ out) {        // [T][H] fp32
  __shared__ __bf16 la[2][BM][LW];
  __shared__ __bf16 lw[2][BNT][LW];
  __shared__ float  lcw[BM][NEXP];

  const int m0   = blockIdx.y * BM;
  const int h0   = blockIdx.x * BNT;
  const int tid  = threadIdx.x;
  const int wid  = tid >> 5;
  const int lane = tid & 31;
  const int half = lane >> 4;
  const int ml   = lane & 15;
  const int wm   = (wid & 3) * 16;
  const int wn   = (wid >> 2) * 32;

  for (int i = tid; i < BM * NEXP; i += 256) {
    lcw[i >> 4][i & 15] = cw[(size_t)(m0 + (i >> 4)) * NEXP + (i & 15)];
  }

  const int crow = tid >> 3;
  const int ccg  = (tid & 7) * 8;

  constexpr int SDOWN = INTER / BK;     // 8 stages per expert
  constexpr int TOTAL = NEXP * SDOWN;   // 128

  auto issue = [&](int idx) {
    const int e  = idx >> 3;
    const int k0 = (idx & (SDOWN - 1)) * BK;
    const int b  = idx & 1;
    const __bf16* ae = act + (size_t)e * TOKENS * INTER + (size_t)m0 * INTER;
    const __bf16* we = wdt + (size_t)e * HID * INTER + (size_t)h0 * INTER;
#pragma unroll
    for (int j = 0; j < 2; ++j) {       // 64x64 tiles: 2 chunks/thread each
      const int row = crow + j * 32;
      async_copy16(lds_lo(&la[b][row][ccg]), ae + (size_t)row * INTER + k0 + ccg);
      async_copy16(lds_lo(&lw[b][row][ccg]), we + (size_t)row * INTER + k0 + ccg);
    }
  };

  v8f acc[2] = {};
  v8f ce[2]  = {};

  issue(0);
  for (int idx = 0; idx < TOTAL; ++idx) {
    async_wait0();
    __syncthreads();
    if (idx + 1 < TOTAL) issue(idx + 1);

    const int b = idx & 1;
    const int s = idx & (SDOWN - 1);
    if (s == 0) {
#pragma unroll
      for (int nf = 0; nf < 2; ++nf)
#pragma unroll
        for (int r = 0; r < 8; ++r) ce[nf][r] = 0.f;
    }
#pragma unroll
    for (int kk = 0; kk < 2; ++kk) {
      const int ko = 32 * kk;
      ABFrag a;
      a.h[0] = *(const v8bf*)&la[b][wm + ml][ko + 8 * half];
      a.h[1] = *(const v8bf*)&la[b][wm + ml][ko + 16 + 8 * half];
#pragma unroll
      for (int nf = 0; nf < 2; ++nf) {
        const int nr = wn + nf * 16 + ml;
        ABFrag bb;
        bb.h[0] = *(const v8bf*)&lw[b][nr][ko + 8 * half];
        bb.h[1] = *(const v8bf*)&lw[b][nr][ko + 16 + 8 * half];
        ce[nf] = __builtin_amdgcn_wmma_f32_16x16x32_bf16(
            false, a.v, false, bb.v, (short)0, ce[nf], false, false);
      }
    }
    if (s == SDOWN - 1) {
      const int e = idx >> 3;
#pragma unroll
      for (int nf = 0; nf < 2; ++nf)
#pragma unroll
        for (int r = 0; r < 8; ++r)
          acc[nf][r] += ce[nf][r] * lcw[wm + r + 8 * half][e];
    }
  }

  // epilogue: routed * 1.0 + zero_w[t] * x[t][h]   (ROUTED_SCALING == 1.0)
#pragma unroll
  for (int nf = 0; nf < 2; ++nf) {
#pragma unroll
    for (int r = 0; r < 8; ++r) {
      int m = m0 + wm + r + 8 * half;
      int h = h0 + wn + nf * 16 + ml;
      out[(size_t)m * HID + h] = acc[nf][r] + zw[m] * x[(size_t)m * HID + h];
    }
  }
}

// ---------------------------------------------------------------------------
extern "C" void kernel_launch(void* const* d_in, const int* in_sizes, int n_in,
                              void* d_out, int out_size, void* d_ws, size_t ws_size,
                              hipStream_t stream) {
  (void)in_sizes; (void)n_in; (void)out_size; (void)ws_size;
  const float* x    = (const float*)d_in[0];
  const float* rw   = (const float*)d_in[1];
  const float* bias = (const float*)d_in[2];
  const float* wg   = (const float*)d_in[3];
  const float* wu   = (const float*)d_in[4];
  const float* wd   = (const float*)d_in[5];
  float* out = (float*)d_out;

  char* p = (char*)d_ws;
  __bf16* xb  = (__bf16*)p; p += (size_t)TOKENS * HID * 2;
  __bf16* wgt = (__bf16*)p; p += (size_t)NEXP * HID * INTER * 2;  // [E][I][H]
  __bf16* wut = (__bf16*)p; p += (size_t)NEXP * HID * INTER * 2;  // [E][I][H]
  __bf16* wdt = (__bf16*)p; p += (size_t)NEXP * INTER * HID * 2;  // [E][H][I]
  __bf16* act = (__bf16*)p; p += (size_t)NEXP * TOKENS * INTER * 2;
  float* cw = (float*)p;    p += (size_t)TOKENS * NEXP * 4;
  float* zw = (float*)p;    p += (size_t)TOKENS * 4;

  const int nx = TOKENS * HID;
  f32_to_bf16_kernel<<<(nx / 4 + 255) / 256, 256, 0, stream>>>(x, xb, nx);

  // Weights: convert + transpose to K-contiguous B layout in one pass.
  transpose_f32_to_bf16<<<dim3(INTER / 32, HID / 32, NEXP), 256, 0, stream>>>(
      wg, wgt, HID, INTER);
  transpose_f32_to_bf16<<<dim3(INTER / 32, HID / 32, NEXP), 256, 0, stream>>>(
      wu, wut, HID, INTER);
  transpose_f32_to_bf16<<<dim3(HID / 32, INTER / 32, NEXP), 256, 0, stream>>>(
      wd, wdt, INTER, HID);

  router_kernel<<<TOKENS / 8, 256, 0, stream>>>(x, rw, bias, cw, zw);

  moe_gateup_kernel<<<dim3(INTER / BNT, TOKENS / BM, NEXP), 256, 0, stream>>>(
      xb, wgt, wut, act);

  moe_down_kernel<<<dim3(HID / BNT, TOKENS / BM), 256, 0, stream>>>(
      act, wdt, cw, zw, x, out);
}
